// PriorAdaptedGaussianFields_34076270526626
// MI455X (gfx1250) — matrix operand order repaired and checked by reading, not compiled, not run on hardware
//
#include <hip/hip_runtime.h>
#include <hip/hip_bf16.h>

typedef __attribute__((ext_vector_type(16))) _Float16 v16h;
typedef __attribute__((ext_vector_type(8)))  float    v8f;
typedef int v4i __attribute__((vector_size(16)));     // matches builtin param spelling

#define NUM_ENT     512     // 8*64 entities
#define NPTS        1000
#define MTILE       32      // points per tile (2 WMMA M-tiles)
#define NTILES      32      // ceil(1000/32)
#define EMB         1024
#define HID         256
#define STY         64

// ---- LDS layout (bytes), all 16B-aligned, f16 row strides padded (+8) ----
#define W2S   264           // f16 stride for 256-col matrices (528B -> bank rotation)
#define W3S   40            // f16 stride for padded 32-col W3
#define PARS  32            // f32 stride for params tile
#define OFF_W2   0                               // 256*264*2 = 135168
#define OFF_W3   135168                          // 256*40*2  =  20480
#define OFF_H1   (OFF_W3 + 20480)                // 32*264*2  =  16896
#define OFF_H2   (OFF_H1 + 16896)                // 32*264*2  =  16896
#define OFF_PAR  (OFF_H2 + 16896)                // 32*32*4   =   4096
#define OFF_ENT  (OFF_PAR + 4096)                // 256*4     =   1024
#define OFF_WP   (OFF_ENT + 1024)                // 3*256*4   =   3072
#define OFF_B2   (OFF_WP + 3072)                 // 256*4     =   1024
#define OFF_B3   (OFF_B2 + 1024)                 // 32*4      =    128
#define SMEM_BYTES (OFF_B3 + 128)                // = 198784  (< 320 KB WGP LDS)

// workspace layout (bytes)
#define WS_ENT   0                               // 512*256 f32 = 524288
#define WS_W2    524288                          // 256*256 f16 = 131072
#define WS_W3    655360                          // 256*32  f16 =  16384

#if defined(__gfx1250__) && __has_builtin(__builtin_amdgcn_global_load_async_to_lds_b128)
#define USE_ASYNC_LDS 1
typedef __attribute__((address_space(1))) v4i GV4i;   // global int4
typedef __attribute__((address_space(3))) v4i LV4i;   // LDS int4
#endif

// ---------------------------------------------------------------------------
// Kernel 1a: per-entity style encoder -> ent_part (512 x 256 f32) in workspace
// ---------------------------------------------------------------------------
__global__ void __launch_bounds__(256)
ent_part_kernel(const float* __restrict__ ef, const int* __restrict__ cls,
                const float* __restrict__ se_w1, const float* __restrict__ se_b1,
                const float* __restrict__ se_g,  const float* __restrict__ se_b,
                const float* __restrict__ se_w2, const float* __restrict__ se_b2,
                const float* __restrict__ gep_w1, const float* __restrict__ gep_b1,
                float* __restrict__ ent_out) {
  __shared__ float efr[EMB];
  __shared__ float red[256];
  __shared__ float hr[HID];
  __shared__ float styl[STY];
  const int e = blockIdx.x, tid = threadIdx.x;

  for (int i = tid; i < EMB; i += 256) efr[i] = ef[e * EMB + i];
  __syncthreads();

  float h = se_b1[tid];
  for (int k = 0; k < EMB; ++k) h += efr[k] * se_w1[k * HID + tid];

  red[tid] = h; __syncthreads();
  for (int s = 128; s > 0; s >>= 1) { if (tid < s) red[tid] += red[tid + s]; __syncthreads(); }
  const float mean = red[0] * (1.0f / 256.0f);
  __syncthreads();
  const float d = h - mean;
  red[tid] = d * d; __syncthreads();
  for (int s = 128; s > 0; s >>= 1) { if (tid < s) red[tid] += red[tid + s]; __syncthreads(); }
  const float var = red[0] * (1.0f / 256.0f);
  __syncthreads();

  const float hn = d * rsqrtf(var + 1e-5f) * se_g[tid] + se_b[tid];
  hr[tid] = hn > 0.f ? hn : 0.f;
  __syncthreads();

  if (tid < STY) {
    float s = se_b2[tid];
    for (int k = 0; k < HID; ++k) s += hr[k] * se_w2[k * STY + tid];
    styl[tid] = s;
  }
  __syncthreads();

  const int cl = cls[e];
  float ep = gep_b1[tid] + gep_w1[(STY + 3 + cl) * HID + tid];   // one-hot @ w_c
  for (int s = 0; s < STY; ++s) ep += styl[s] * gep_w1[s * HID + tid];
  ent_out[e * HID + tid] = ep;
}

// ---------------------------------------------------------------------------
// Kernel 1b: one-time f32 -> f16 weight conversion into workspace
// (async global->LDS copies move raw bytes, so conversion happens up front)
// ---------------------------------------------------------------------------
__global__ void __launch_bounds__(256)
prep_weights_kernel(const float* __restrict__ w2, const float* __restrict__ w3,
                    _Float16* __restrict__ w2o, _Float16* __restrict__ w3o) {
  const int stride = gridDim.x * 256;
  for (int i = blockIdx.x * 256 + threadIdx.x; i < HID * HID; i += stride)
    w2o[i] = (_Float16)w2[i];
  for (int i = blockIdx.x * 256 + threadIdx.x; i < HID * 32; i += stride) {
    const int k = i >> 5, n = i & 31;
    w3o[i] = (_Float16)((n < 19) ? w3[k * 19 + n] : 0.0f);
  }
}

// ---------------------------------------------------------------------------
// wave32 V_WMMA_F32_16X16X32_F16 fragment helpers (layouts per ISA 7.12.2)
// ---------------------------------------------------------------------------
union frag16 { v16h v; uint4 u[2]; };

__device__ __forceinline__ v16h load_a_frag(const _Float16* row) {
  frag16 f;
  f.u[0] = *(const uint4*)(row);        // K = h*8 .. h*8+7
  f.u[1] = *(const uint4*)(row + 16);   // K = 16+h*8 .. +7
  return f.v;
}
__device__ __forceinline__ v16h load_b_frag(const _Float16* col, int ld) {
  v16h v;
  #pragma unroll
  for (int s = 0; s < 16; ++s) v[s] = col[(s < 8 ? s : s + 8) * ld];
  return v;
}
#define WMMA_F16(A, B, C) \
  __builtin_amdgcn_wmma_f32_16x16x32_f16(false, (A), false, (B), (short)0, (C), false, false)

// ---------------------------------------------------------------------------
// Kernel 2: fused point-MLP. One block per entity, 8 waves, W2 LDS-resident.
// 32 tiles of 32 points; each wave computes a 32x32 C block (2x2 WMMA tiles).
// ---------------------------------------------------------------------------
__global__ void __launch_bounds__(256)
gauss_field_kernel(const float* __restrict__ ent_ws, const int* __restrict__ adapter_ids,
                   const float* __restrict__ cp,
                   const float* __restrict__ gep_w1,
                   const _Float16* __restrict__ ws_w2, const _Float16* __restrict__ ws_w3,
                   const float* __restrict__ gep_b2, const float* __restrict__ gep_b3,
                   const float* __restrict__ mu_tab, const float* __restrict__ cov_tab,
                   const float* __restrict__ op_tab, const float* __restrict__ sh_tab,
                   float* __restrict__ out_mu, float* __restrict__ out_cov,
                   float* __restrict__ out_op, float* __restrict__ out_sh) {
  extern __shared__ char smem[];
  _Float16* w2s  = (_Float16*)(smem + OFF_W2);
  _Float16* w3s  = (_Float16*)(smem + OFF_W3);
  _Float16* h1s  = (_Float16*)(smem + OFF_H1);
  _Float16* h2s  = (_Float16*)(smem + OFF_H2);
  float*    pars = (float*)   (smem + OFF_PAR);
  float*    entp = (float*)   (smem + OFF_ENT);
  float*    wps  = (float*)   (smem + OFF_WP);
  float*    b2s  = (float*)   (smem + OFF_B2);
  float*    b3s  = (float*)   (smem + OFF_B3);

  const int e = blockIdx.x, tid = threadIdx.x;
  const int wave = tid >> 5, lane = tid & 31;
  const int r = lane & 15, hh = lane >> 4;
  const int aid = adapter_ids[e];

  // ---- stage f16 weights into LDS ----
#if defined(USE_ASYNC_LDS)
  // GLOBAL_LOAD_ASYNC_TO_LDS_B128 (ASYNCcnt-tracked, no VGPR data movement):
  // 8192 x 16B chunks for W2 (264-f16 padded LDS rows), 1024 chunks for W3.
  for (int i = tid; i < HID * 32; i += 256) {
    const int k = i >> 5, c = (i & 31) * 8;
    __builtin_amdgcn_global_load_async_to_lds_b128(
        (GV4i*)(ws_w2 + k * HID + c), (LV4i*)(w2s + k * W2S + c), 0, 0);
  }
  for (int i = tid; i < HID * 4; i += 256) {
    const int k = i >> 2, c = (i & 3) * 8;
    __builtin_amdgcn_global_load_async_to_lds_b128(
        (GV4i*)(ws_w3 + k * 32 + c), (LV4i*)(w3s + k * W3S + c), 0, 0);
  }
#if __has_builtin(__builtin_amdgcn_s_wait_asynccnt)
  __builtin_amdgcn_s_wait_asynccnt(0);
#else
  asm volatile("s_wait_asynccnt 0x0" ::: "memory");
#endif
#else
  for (int i = tid; i < HID * 32; i += 256) {
    const int k = i >> 5, c = (i & 31) * 8;
    *(uint4*)(w2s + k * W2S + c) = *(const uint4*)(ws_w2 + k * HID + c);
  }
  for (int i = tid; i < HID * 4; i += 256) {
    const int k = i >> 2, c = (i & 3) * 8;
    *(uint4*)(w3s + k * W3S + c) = *(const uint4*)(ws_w3 + k * 32 + c);
  }
#endif
  b2s[tid]  = gep_b2[tid];
  entp[tid] = ent_ws[e * HID + tid];
  wps[0 * HID + tid] = gep_w1[(STY + 0) * HID + tid];
  wps[1 * HID + tid] = gep_w1[(STY + 1) * HID + tid];
  wps[2 * HID + tid] = gep_w1[(STY + 2) * HID + tid];
  if (tid < 32) b3s[tid] = (tid < 19) ? gep_b3[tid] : 0.0f;
  __syncthreads();

  for (int tile = 0; tile < NTILES; ++tile) {
    // ---- build h1 tile (32 x 256 f16) + zero params accumulator ----
    {
      const int c0 = (tid & 15) * 16;
      #pragma unroll
      for (int rr = 0; rr < 2; ++rr) {
        const int p = (tid >> 4) + 16 * rr;
        const int pg = tile * MTILE + p;
        float cx = 0.f, cy = 0.f, cz = 0.f;
        if (pg < NPTS) { cx = cp[pg * 3 + 0]; cy = cp[pg * 3 + 1]; cz = cp[pg * 3 + 2]; }
        #pragma unroll
        for (int j = 0; j < 16; ++j) {
          const int c = c0 + j;
          float v = entp[c] + cx * wps[c] + cy * wps[HID + c] + cz * wps[2 * HID + c];
          h1s[p * W2S + c] = (_Float16)(v > 0.f ? v : 0.f);
        }
      }
      pars[tid] = 0.f; pars[256 + tid] = 0.f;
      pars[512 + tid] = 0.f; pars[768 + tid] = 0.f;       // 32x32 f32
    }
    __syncthreads();

    // ---- GEMM2: h2(32x256) = relu(h1 @ W2 + b2)
    // wave w -> N columns [w*32, w*32+32); 2x2 register C tile, shared frags
    {
      const int n0 = wave * 32;
      v8f acc00 = {}, acc01 = {}, acc10 = {}, acc11 = {};
      #pragma unroll
      for (int k0 = 0; k0 < HID; k0 += 32) {
        const v16h a0 = load_a_frag(&h1s[r * W2S + k0 + hh * 8]);
        const v16h a1 = load_a_frag(&h1s[(16 + r) * W2S + k0 + hh * 8]);
        const v16h b0 = load_b_frag(&w2s[(k0 + hh * 8) * W2S + n0 + r], W2S);
        const v16h b1 = load_b_frag(&w2s[(k0 + hh * 8) * W2S + n0 + 16 + r], W2S);
        acc00 = WMMA_F16(a0, b0, acc00);
        acc10 = WMMA_F16(a1, b0, acc10);
        acc01 = WMMA_F16(a0, b1, acc01);
        acc11 = WMMA_F16(a1, b1, acc11);
      }
      const float bias0 = b2s[n0 + r], bias1 = b2s[n0 + 16 + r];
      #pragma unroll
      for (int j = 0; j < 8; ++j) {                 // C row = hh*8 + j (+16 for m-tile 1)
        const int row = hh * 8 + j;
        float v;
        v = acc00[j] + bias0; h2s[row * W2S + n0 + r]             = (_Float16)(v > 0.f ? v : 0.f);
        v = acc10[j] + bias0; h2s[(16 + row) * W2S + n0 + r]      = (_Float16)(v > 0.f ? v : 0.f);
        v = acc01[j] + bias1; h2s[row * W2S + n0 + 16 + r]        = (_Float16)(v > 0.f ? v : 0.f);
        v = acc11[j] + bias1; h2s[(16 + row) * W2S + n0 + 16 + r] = (_Float16)(v > 0.f ? v : 0.f);
      }
    }
    __syncthreads();

    // ---- GEMM3: params(32x32) += h2 @ W3pad ; K=256 split across 8 waves
    {
      const int k0 = wave * 32;
      const v16h a0 = load_a_frag(&h2s[r * W2S + k0 + hh * 8]);
      const v16h a1 = load_a_frag(&h2s[(16 + r) * W2S + k0 + hh * 8]);
      #pragma unroll
      for (int t = 0; t < 2; ++t) {
        const int n0 = t * 16;
        const v16h b = load_b_frag(&w3s[(k0 + hh * 8) * W3S + n0 + r], W3S);
        v8f acc0 = {}, acc1 = {};
        acc0 = WMMA_F16(a0, b, acc0);
        acc1 = WMMA_F16(a1, b, acc1);
        #pragma unroll
        for (int j = 0; j < 8; ++j) {
          atomicAdd(&pars[(hh * 8 + j) * PARS + n0 + r], acc0[j]);        // ds_add_f32
          atomicAdd(&pars[(16 + hh * 8 + j) * PARS + n0 + r], acc1[j]);
        }
      }
    }
    __syncthreads();

    // ---- epilogue: bias, sigmoid(ch 9), adapter gather, scatter to outputs
    for (int i = tid; i < MTILE * 19; i += 256) {
      const int p = i / 19, c = i - p * 19;
      const int pg = tile * MTILE + p;
      if (pg < NPTS) {
        const float v   = pars[p * PARS + c] + b3s[c];
        const int idx   = e * NPTS + pg;
        const int tbase = aid * NPTS + pg;
        if (c < 3)       out_mu[idx * 3 + c]        = v + mu_tab[tbase * 3 + c];
        else if (c < 9)  out_cov[idx * 6 + (c - 3)] = v + cov_tab[tbase * 6 + (c - 3)];
        else if (c == 9) out_op[idx] = 1.0f / (1.0f + expf(-v)) + op_tab[tbase];
        else             out_sh[idx * 9 + (c - 10)] = v + sh_tab[tbase * 9 + (c - 10)];
      }
    }
    __syncthreads();
  }
}

// ---------------------------------------------------------------------------
extern "C" void kernel_launch(void* const* d_in, const int* in_sizes, int n_in,
                              void* d_out, int out_size, void* d_ws, size_t ws_size,
                              hipStream_t stream) {
  const float* ef       = (const float*)d_in[0];
  const int*   cls      = (const int*)  d_in[1];
  const int*   aid      = (const int*)  d_in[2];
  const float* se_w1    = (const float*)d_in[3];
  const float* se_b1    = (const float*)d_in[4];
  const float* se_g     = (const float*)d_in[5];
  const float* se_b     = (const float*)d_in[6];
  const float* se_w2    = (const float*)d_in[7];
  const float* se_b2    = (const float*)d_in[8];
  const float* cp       = (const float*)d_in[9];
  const float* gep_w1   = (const float*)d_in[10];
  const float* gep_b1   = (const float*)d_in[11];
  const float* gep_w2   = (const float*)d_in[12];
  const float* gep_b2   = (const float*)d_in[13];
  const float* gep_w3   = (const float*)d_in[14];
  const float* gep_b3   = (const float*)d_in[15];
  const float* mu_tab   = (const float*)d_in[16];
  const float* cov_tab  = (const float*)d_in[17];
  const float* op_tab   = (const float*)d_in[18];
  const float* sh_tab   = (const float*)d_in[19];

  char* ws = (char*)d_ws;
  float*     ent_ws = (float*)(ws + WS_ENT);
  _Float16*  w2_ws  = (_Float16*)(ws + WS_W2);
  _Float16*  w3_ws  = (_Float16*)(ws + WS_W3);

  float* out   = (float*)d_out;
  float* o_mu  = out;                               // (512,1000,3)
  float* o_cov = out + NUM_ENT * NPTS * 3;          // (512,1000,6)
  float* o_op  = o_cov + NUM_ENT * NPTS * 6;        // (512,1000,1)
  float* o_sh  = o_op + NUM_ENT * NPTS;             // (512,1000,9)

  ent_part_kernel<<<NUM_ENT, 256, 0, stream>>>(ef, cls, se_w1, se_b1, se_g, se_b,
                                               se_w2, se_b2, gep_w1, gep_b1, ent_ws);
  prep_weights_kernel<<<64, 256, 0, stream>>>(gep_w2, gep_w3, w2_ws, w3_ws);

  (void)hipFuncSetAttribute(reinterpret_cast<const void*>(gauss_field_kernel),
                            hipFuncAttributeMaxDynamicSharedMemorySize, SMEM_BYTES);
  gauss_field_kernel<<<NUM_ENT, 256, SMEM_BYTES, stream>>>(
      ent_ws, aid, cp, gep_w1, w2_ws, w3_ws, gep_b2, gep_b3,
      mu_tab, cov_tab, op_tab, sh_tab, o_mu, o_cov, o_op, o_sh);
}
